// LightningBlock_64450279244022
// MI455X (gfx1250) — matrix operand order repaired and checked by reading, not compiled, not run on hardware
//
#include <hip/hip_runtime.h>
#include <hip/hip_bf16.h>
#include <stdint.h>

// ---------------------------------------------------------------------------
// Problem constants (B=2, N=4096, D=2048, BS=256)
// ---------------------------------------------------------------------------
#define BATCH   2
#define SEQ     4096
#define DMODEL  2048
#define CHUNK   256
#define NCHUNK  (SEQ / CHUNK)

// GEMM tiling: block = 128x128, K-step 32, 8 waves (wave32), wave tile 64x32
#define BM 128
#define BN 128
#define BK 32
#define LDS_STRIDE 40  // BK + 8 bf16 padding
#define TILE_ELEMS (BM * LDS_STRIDE)

typedef __attribute__((ext_vector_type(16))) __bf16 v16bf;
typedef __attribute__((ext_vector_type(8)))  __bf16 v8bf;
typedef __attribute__((ext_vector_type(8)))  float  v8f;
typedef __attribute__((ext_vector_type(4)))  float  v4f;
typedef __attribute__((ext_vector_type(4)))  int    v4i;

// ---------------------------------------------------------------------------
// Async global->LDS copy (CDNA5 GLOBAL_LOAD_ASYNC_TO_LDS_B128, ASYNCcnt).
// Builtin signature (from compiler diagnostic): (v4i __device__*, v4i __shared__*, imm, imm)
// Falls back to load+ds_store if the builtin is unavailable.
// ---------------------------------------------------------------------------
#if defined(__has_builtin)
#  if __has_builtin(__builtin_amdgcn_global_load_async_to_lds_b128) && \
      __has_builtin(__builtin_amdgcn_s_wait_asynccnt)
#    define HAS_ASYNC_LDS 1
#  endif
#endif
#ifndef HAS_ASYNC_LDS
#  define HAS_ASYNC_LDS 0
#endif

static __device__ inline void copy16_g2l(const __bf16* g, __bf16* l) {
#if HAS_ASYNC_LDS
    typedef __attribute__((address_space(1))) v4i* gptr_t;
    typedef __attribute__((address_space(3))) v4i* lptr_t;
    __builtin_amdgcn_global_load_async_to_lds_b128(
        (gptr_t)(uintptr_t)g, (lptr_t)(uint32_t)(uintptr_t)l, 0, 0);
#else
    *(v8bf*)l = *(const v8bf*)g;
#endif
}
static __device__ inline void lds_fence_async() {
#if HAS_ASYNC_LDS
    __builtin_amdgcn_s_wait_asynccnt(0);
#endif
}

// ---------------------------------------------------------------------------
// WMMA microkernel: one BK=32 K-step of an 8-wave 128x128 block tile.
// As/Bs: [128][LDS_STRIDE] bf16, rows = M/N, fast dim = K.
// A frag (16x32): lane = m + 16h holds K in [8h,8h+8) and [16+8h,16+8h+8)
// B frag (32x16): lane = n + 16h holds K in [8h,8h+8) and [16+8h,16+8h+8)
// ---------------------------------------------------------------------------
static __device__ inline void wmma_step(const __bf16* As, const __bf16* Bs,
                                        int wm, int wn, int lane,
                                        v8f (*acc)[2]) {
    const int m16 = lane & 15;
    const int h   = lane >> 4;
    v16bf a[4];
    v16bf b[2];
#pragma unroll
    for (int i = 0; i < 4; ++i) {
        const __bf16* p = As + (wm * 64 + i * 16 + m16) * LDS_STRIDE + h * 8;
        v8bf lo = *(const v8bf*)p;
        v8bf hi = *(const v8bf*)(p + 16);
        a[i] = __builtin_shufflevector(lo, hi, 0, 1, 2, 3, 4, 5, 6, 7,
                                       8, 9, 10, 11, 12, 13, 14, 15);
    }
#pragma unroll
    for (int j = 0; j < 2; ++j) {
        const __bf16* p = Bs + (wn * 32 + j * 16 + m16) * LDS_STRIDE + h * 8;
        v8bf lo = *(const v8bf*)p;
        v8bf hi = *(const v8bf*)(p + 16);
        b[j] = __builtin_shufflevector(lo, hi, 0, 1, 2, 3, 4, 5, 6, 7,
                                       8, 9, 10, 11, 12, 13, 14, 15);
    }
#pragma unroll
    for (int i = 0; i < 4; ++i)
#pragma unroll
        for (int j = 0; j < 2; ++j)
            acc[i][j] = __builtin_amdgcn_wmma_f32_16x16x32_bf16(
                false, a[i], false, b[j], (short)0, acc[i][j], false, false);
}

// ---------------------------------------------------------------------------
// Double-buffered GEMM pipeline, both operands bf16 contiguous-along-K.
// smem must hold 4 tiles (As0,Bs0,As1,Bs1).  One barrier per K-step.
// ---------------------------------------------------------------------------
static __device__ inline void gemm_pipe_bf(const __bf16* __restrict__ A, int lda,
                                           int arow0, int acol0,
                                           const __bf16* __restrict__ B, int ldb,
                                           int brow0, int bcol0,
                                           int Ktot, __bf16* smem,
                                           int tid, int wm, int wn, int lane,
                                           v8f (*acc)[2]) {
    __bf16* As[2] = { smem,                  smem + 2 * TILE_ELEMS };
    __bf16* Bs[2] = { smem + TILE_ELEMS,     smem + 3 * TILE_ELEMS };
    auto stage = [&](int k0, __bf16* as, __bf16* bs) {
#pragma unroll
        for (int u = 0; u < 2; ++u) {
            const int c = tid + u * 256;
            const int r = c >> 2, col = (c & 3) << 3;
            copy16_g2l(&A[(size_t)(arow0 + r) * lda + acol0 + k0 + col],
                       &as[r * LDS_STRIDE + col]);
            copy16_g2l(&B[(size_t)(brow0 + r) * ldb + bcol0 + k0 + col],
                       &bs[r * LDS_STRIDE + col]);
        }
    };
    int p = 0;
    stage(0, As[0], Bs[0]);
    for (int k0 = 0; k0 < Ktot; k0 += BK) {
        lds_fence_async();
        __syncthreads();
        if (k0 + BK < Ktot) stage(k0 + BK, As[p ^ 1], Bs[p ^ 1]);
        wmma_step(As[p], Bs[p], wm, wn, lane, acc);
        p ^= 1;
    }
}

// Variant with fp32 B operand (KV^T state), converted on stage with native cvt.
static __device__ inline void gemm_pipe_f32B(const __bf16* __restrict__ A, int lda,
                                             int arow0, int acol0,
                                             const float* __restrict__ B, int ldb,
                                             int brow0, int bcol0,
                                             int Ktot, __bf16* smem,
                                             int tid, int wm, int wn, int lane,
                                             v8f (*acc)[2]) {
    __bf16* As[2] = { smem,                  smem + 2 * TILE_ELEMS };
    __bf16* Bs[2] = { smem + TILE_ELEMS,     smem + 3 * TILE_ELEMS };
    auto stage = [&](int k0, __bf16* as, __bf16* bs) {
#pragma unroll
        for (int u = 0; u < 2; ++u) {
            const int c = tid + u * 256;
            const int r = c >> 2, col = (c & 3) << 3;
            copy16_g2l(&A[(size_t)(arow0 + r) * lda + acol0 + k0 + col],
                       &as[r * LDS_STRIDE + col]);
            const float* bp = &B[(size_t)(brow0 + r) * ldb + bcol0 + k0 + col];
            v4f lo = *(const v4f*)bp;
            v4f hi = *(const v4f*)(bp + 4);
            v8bf t;
#pragma unroll
            for (int i = 0; i < 4; ++i) { t[i] = (__bf16)lo[i]; t[i + 4] = (__bf16)hi[i]; }
            *(v8bf*)&bs[r * LDS_STRIDE + col] = t;
        }
    };
    int p = 0;
    stage(0, As[0], Bs[0]);
    for (int k0 = 0; k0 < Ktot; k0 += BK) {
        lds_fence_async();
        __syncthreads();
        if (k0 + BK < Ktot) stage(k0 + BK, As[p ^ 1], Bs[p ^ 1]);
        wmma_step(As[p], Bs[p], wm, wn, lane, acc);
        p ^= 1;
    }
}

// ---------------------------------------------------------------------------
// Kernel: RMS norm fp32 -> bf16.  One 256-thread block per row of 2048.
// ---------------------------------------------------------------------------
__global__ __launch_bounds__(256) void k_rmsnorm(const float* __restrict__ x,
                                                 __bf16* __restrict__ xn) {
    const int row = blockIdx.x;
    const float* xr = x + (size_t)row * DMODEL;
    __bf16* outr = xn + (size_t)row * DMODEL;
    const int base = threadIdx.x * 8;
    v4f a = *(const v4f*)(xr + base);
    v4f b = *(const v4f*)(xr + base + 4);
    float s = 0.f;
#pragma unroll
    for (int i = 0; i < 4; ++i) { s += a[i] * a[i]; s += b[i] * b[i]; }
    __shared__ float red[256];
    red[threadIdx.x] = s;
    __syncthreads();
#pragma unroll
    for (int off = 128; off > 0; off >>= 1) {
        if ((int)threadIdx.x < off) red[threadIdx.x] += red[threadIdx.x + off];
        __syncthreads();
    }
    const float scale = rsqrtf(red[0] * (1.0f / (float)DMODEL) + 1.1920929e-7f);
    v8bf o;
#pragma unroll
    for (int i = 0; i < 4; ++i) {
        o[i]     = (__bf16)(a[i] * scale);
        o[i + 4] = (__bf16)(b[i] * scale);
    }
    *(v8bf*)(outr + base) = o;
}

// ---------------------------------------------------------------------------
// Kernel: fp32 -> bf16 bulk convert (weights).  8 elems / thread.
// ---------------------------------------------------------------------------
__global__ __launch_bounds__(256) void k_w2bf(const float* __restrict__ w,
                                              __bf16* __restrict__ o) {
    const size_t i = ((size_t)blockIdx.x * 256 + threadIdx.x) * 8;
    v4f a = *(const v4f*)(w + i);
    v4f b = *(const v4f*)(w + i + 4);
    v8bf t;
#pragma unroll
    for (int j = 0; j < 4; ++j) { t[j] = (__bf16)a[j]; t[j + 4] = (__bf16)b[j]; }
    *(v8bf*)(o + i) = t;
}

// ---------------------------------------------------------------------------
// Kernel: bf16 transpose [SEQ,D] -> [D,SEQ] per batch (64x64 LDS tiles).
// grid = (SEQ/64, D/64, BATCH)
// ---------------------------------------------------------------------------
__global__ __launch_bounds__(256) void k_transpose(const __bf16* __restrict__ in,
                                                   __bf16* __restrict__ out) {
    __shared__ __bf16 tile[64 * 80];  // [d][n], stride 80 keeps b128 reads aligned
    const int b = blockIdx.z;
    const __bf16* ib = in + (size_t)b * SEQ * DMODEL;
    __bf16* ob = out + (size_t)b * DMODEL * SEQ;
    const int n0 = blockIdx.x * 64, d0 = blockIdx.y * 64;
    const int tid = threadIdx.x;
#pragma unroll
    for (int u = 0; u < 2; ++u) {
        const int c = tid + u * 256;
        const int r = c >> 3, col = (c & 7) << 3;         // n-row, d-col
        v8bf v = *(const v8bf*)&ib[(size_t)(n0 + r) * DMODEL + d0 + col];
#pragma unroll
        for (int i = 0; i < 8; ++i) tile[(col + i) * 80 + r] = v[i];
    }
    __syncthreads();
#pragma unroll
    for (int u = 0; u < 2; ++u) {
        const int c = tid + u * 256;
        const int rd = c >> 3, ncol = (c & 7) << 3;       // d-row, n-col
        v8bf v = *(const v8bf*)&tile[rd * 80 + ncol];
        *(v8bf*)&ob[(size_t)(d0 + rd) * SEQ + n0 + ncol] = v;
    }
}

// ---------------------------------------------------------------------------
// Kernel: C = act(A[M,K]bf16 @ B[N,K]bf16^T)
// ACT: 0 = silu -> bf16, 1 = sigmoid -> bf16, 2 = none -> fp32
// grid = (N/BN, M/BM)
// ---------------------------------------------------------------------------
template <int ACT>
__global__ __launch_bounds__(256) void k_gemm_nt(const __bf16* __restrict__ A,
                                                 const __bf16* __restrict__ B,
                                                 void* __restrict__ C,
                                                 int M, int N, int K) {
    __shared__ __bf16 smem[4 * TILE_ELEMS];
    const int tid = threadIdx.x, lane = tid & 31, wave = tid >> 5;
    const int wm = wave & 1, wn = wave >> 1;
    const int bm0 = blockIdx.y * BM, bn0 = blockIdx.x * BN;
    v8f acc[4][2] = {};
    gemm_pipe_bf(A, K, bm0, 0, B, K, bn0, 0, K, smem, tid, wm, wn, lane, acc);

    const int m16 = lane & 15, h = lane >> 4;
#pragma unroll
    for (int i = 0; i < 4; ++i) {
        const int rbase = bm0 + wm * 64 + i * 16 + h * 8;
#pragma unroll
        for (int j = 0; j < 2; ++j) {
            const int col = bn0 + wn * 32 + j * 16 + m16;
#pragma unroll
            for (int r = 0; r < 8; ++r) {
                float v = acc[i][j][r];
                const size_t idx = (size_t)(rbase + r) * N + col;
                if (ACT == 0) {
                    ((__bf16*)C)[idx] = (__bf16)(v / (1.f + expf(-v)));
                } else if (ACT == 1) {
                    ((__bf16*)C)[idx] = (__bf16)(1.f / (1.f + expf(-v)));
                } else {
                    ((float*)C)[idx] = v;
                }
            }
        }
    }
}

// ---------------------------------------------------------------------------
// Kernel: masked chunk scores  S = tril(Q_t K_t^T), bf16.
// grid = (CHUNK/BN, CHUNK/BM, BATCH)
// ---------------------------------------------------------------------------
__global__ __launch_bounds__(256) void k_scores(const __bf16* __restrict__ q,
                                                const __bf16* __restrict__ kmat,
                                                __bf16* __restrict__ S, int t) {
    __shared__ __bf16 smem[4 * TILE_ELEMS];
    const int tid = threadIdx.x, lane = tid & 31, wave = tid >> 5;
    const int wm = wave & 1, wn = wave >> 1;
    const int b = blockIdx.z;
    const int bm0 = blockIdx.y * BM, bn0 = blockIdx.x * BN;
    const size_t base = ((size_t)b * SEQ + (size_t)t * CHUNK) * DMODEL;
    v8f acc[4][2] = {};
    gemm_pipe_bf(q + base, DMODEL, bm0, 0, kmat + base, DMODEL, bn0, 0,
                 DMODEL, smem, tid, wm, wn, lane, acc);

    const int m16 = lane & 15, h = lane >> 4;
    __bf16* Sb = S + (size_t)b * CHUNK * CHUNK;
#pragma unroll
    for (int i = 0; i < 4; ++i)
#pragma unroll
        for (int j = 0; j < 2; ++j)
#pragma unroll
            for (int r = 0; r < 8; ++r) {
                const int row = bm0 + wm * 64 + i * 16 + h * 8 + r;
                const int col = bn0 + wn * 32 + j * 16 + m16;
                const float v = (col <= row) ? acc[i][j][r] : 0.f;
                Sb[(size_t)row * CHUNK + col] = (__bf16)v;
            }
}

// ---------------------------------------------------------------------------
// Kernel: O = S * V_t + Q_t * KV, gated by g, bf16 out.
// Phase 1: A = S [CHUNK,CHUNK], B = vT[e, t*CHUNK + j]   (all contiguous NT)
// Phase 2: A = Q chunk,         B = KVT[e, d] fp32 (convert on stage)
// grid = (D/BN, CHUNK/BM, BATCH)
// ---------------------------------------------------------------------------
__global__ __launch_bounds__(256) void k_attn_out(const __bf16* __restrict__ S,
                                                  const __bf16* __restrict__ vT,
                                                  const __bf16* __restrict__ q,
                                                  const float* __restrict__ kvt,
                                                  const __bf16* __restrict__ g,
                                                  __bf16* __restrict__ og, int t) {
    __shared__ __bf16 smem[4 * TILE_ELEMS];
    const int tid = threadIdx.x, lane = tid & 31, wave = tid >> 5;
    const int wm = wave & 1, wn = wave >> 1;
    const int b = blockIdx.z;
    const int bm0 = blockIdx.y * BM;   // i-tile within chunk
    const int bn0 = blockIdx.x * BN;   // e-tile within d_model
    const size_t cbase = ((size_t)b * SEQ + (size_t)t * CHUNK) * DMODEL;
    v8f acc[4][2] = {};

    // phase 1: intra-chunk (S masked) * V
    gemm_pipe_bf(S + (size_t)b * CHUNK * CHUNK, CHUNK, bm0, 0,
                 vT + (size_t)b * DMODEL * SEQ, SEQ, bn0, t * CHUNK,
                 CHUNK, smem, tid, wm, wn, lane, acc);
    // phase 2: cross-chunk Q * KV  (KVT[e,d] fp32, rows = e, contraction = d)
    gemm_pipe_f32B(q + cbase, DMODEL, bm0, 0,
                   kvt + (size_t)b * DMODEL * DMODEL, DMODEL, bn0, 0,
                   DMODEL, smem, tid, wm, wn, lane, acc);

    const int m16 = lane & 15, h = lane >> 4;
#pragma unroll
    for (int i = 0; i < 4; ++i)
#pragma unroll
        for (int j = 0; j < 2; ++j)
#pragma unroll
            for (int r = 0; r < 8; ++r) {
                const int ri = bm0 + wm * 64 + i * 16 + h * 8 + r;
                const int ce = bn0 + wn * 32 + j * 16 + m16;
                const size_t idx = cbase + (size_t)ri * DMODEL + ce;
                og[idx] = (__bf16)(acc[i][j][r] * (float)g[idx]);
            }
}

// ---------------------------------------------------------------------------
// Kernel: KVT[e,d] += sum_j V[j,e] K[j,d]  ==  vT-rows x kT-rows (NT GEMM).
// grid = (D/BN (d), D/BM (e), BATCH)
// ---------------------------------------------------------------------------
__global__ __launch_bounds__(256) void k_kvt_update(const __bf16* __restrict__ vT,
                                                    const __bf16* __restrict__ kT,
                                                    float* __restrict__ kvt, int t) {
    __shared__ __bf16 smem[4 * TILE_ELEMS];
    const int tid = threadIdx.x, lane = tid & 31, wave = tid >> 5;
    const int wm = wave & 1, wn = wave >> 1;
    const int b = blockIdx.z;
    const int bm0 = blockIdx.y * BM;   // e-tile
    const int bn0 = blockIdx.x * BN;   // d-tile
    v8f acc[4][2] = {};
    gemm_pipe_bf(vT + (size_t)b * DMODEL * SEQ, SEQ, bm0, t * CHUNK,
                 kT + (size_t)b * DMODEL * SEQ, SEQ, bn0, t * CHUNK,
                 CHUNK, smem, tid, wm, wn, lane, acc);

    float* KVb = kvt + (size_t)b * DMODEL * DMODEL;
    const int m16 = lane & 15, h = lane >> 4;
#pragma unroll
    for (int i = 0; i < 4; ++i)
#pragma unroll
        for (int j = 0; j < 2; ++j)
#pragma unroll
            for (int r = 0; r < 8; ++r) {
                const int re = bm0 + wm * 64 + i * 16 + h * 8 + r;
                const int cd = bn0 + wn * 32 + j * 16 + m16;
                KVb[(size_t)re * DMODEL + cd] += acc[i][j][r];
            }
}

// ---------------------------------------------------------------------------
// Host launcher
// ---------------------------------------------------------------------------
extern "C" void kernel_launch(void* const* d_in, const int* in_sizes, int n_in,
                              void* d_out, int out_size, void* d_ws, size_t ws_size,
                              hipStream_t stream) {
    (void)in_sizes; (void)n_in; (void)out_size; (void)ws_size;
    const float* x  = (const float*)d_in[0];
    const float* W[5] = { (const float*)d_in[1], (const float*)d_in[2],
                          (const float*)d_in[3], (const float*)d_in[4],
                          (const float*)d_in[5] };            // Wq,Wk,Wv,Wg,Wp
    float* out = (float*)d_out;

    const size_t NB = (size_t)BATCH * SEQ * DMODEL;           // 16,777,216 elems
    const size_t DD = (size_t)DMODEL * DMODEL;
    __bf16* xn  = (__bf16*)d_ws;                              // also 'og' (xn dead after projections)
    __bf16* qb  = xn + NB;
    __bf16* kb  = qb + NB;
    __bf16* vb  = kb + NB;
    __bf16* gb  = vb + NB;
    __bf16* kT  = gb + NB;                                    // [B][D][SEQ]
    __bf16* vT  = kT + NB;
    __bf16* wbf = vT + NB;                                    // 5 x [D][D] bf16
    __bf16* sb  = wbf + 5 * DD;                               // [B][CHUNK][CHUNK]
    float*  kvt = (float*)(sb + (size_t)BATCH * CHUNK * CHUNK); // KV^T state fp32

    (void)hipMemsetAsync(kvt, 0, (size_t)BATCH * DD * sizeof(float), stream);

    // 1) RMS norm -> bf16; weights -> bf16
    k_rmsnorm<<<BATCH * SEQ, 256, 0, stream>>>(x, xn);
    for (int w = 0; w < 5; ++w)
        k_w2bf<<<DD / (256 * 8), 256, 0, stream>>>(W[w], wbf + w * DD);

    // 2) projections q,k,v (silu) and g (sigmoid)
    dim3 gg(DMODEL / BN, (BATCH * SEQ) / BM);
    k_gemm_nt<0><<<gg, 256, 0, stream>>>(xn, wbf + 0 * DD, qb, BATCH * SEQ, DMODEL, DMODEL);
    k_gemm_nt<0><<<gg, 256, 0, stream>>>(xn, wbf + 1 * DD, kb, BATCH * SEQ, DMODEL, DMODEL);
    k_gemm_nt<0><<<gg, 256, 0, stream>>>(xn, wbf + 2 * DD, vb, BATCH * SEQ, DMODEL, DMODEL);
    k_gemm_nt<1><<<gg, 256, 0, stream>>>(xn, wbf + 3 * DD, gb, BATCH * SEQ, DMODEL, DMODEL);

    // 3) k^T, v^T for contiguous-K staging in the scan
    dim3 tg(SEQ / 64, DMODEL / 64, BATCH);
    k_transpose<<<tg, 256, 0, stream>>>(kb, kT);
    k_transpose<<<tg, 256, 0, stream>>>(vb, vT);

    // 4) chunked linear-attention scan (sequential, stream-ordered)
    for (int t = 0; t < NCHUNK; ++t) {
        k_scores    <<<dim3(CHUNK / BN, CHUNK / BM, BATCH), 256, 0, stream>>>(qb, kb, sb, t);
        k_attn_out  <<<dim3(DMODEL / BN, CHUNK / BM, BATCH), 256, 0, stream>>>(sb, vT, qb, kvt, gb, xn /*og*/, t);
        k_kvt_update<<<dim3(DMODEL / BN, DMODEL / BM, BATCH), 256, 0, stream>>>(vT, kT, kvt, t);
    }

    // 5) output projection: out = og @ Wp^T (fp32 out)
    k_gemm_nt<2><<<gg, 256, 0, stream>>>(xn /*og*/, wbf + 4 * DD, out, BATCH * SEQ, DMODEL, DMODEL);
}
